// Decoder_50027779064237
// MI455X (gfx1250) — compile-verified, matrix operand order checked by reading, not run on hardware
//
#include <hip/hip_runtime.h>
#include <stdint.h>

// ---------------------------------------------------------------------------
// Shapes
//   B=128, S=1024, ENC2=512 (2*ENC_H), DEC_H=512, RNN_IN=768, OUT_DIM=256
// d_out = [out (128*256) | h_new (128*512)] f32
// ---------------------------------------------------------------------------

typedef __attribute__((ext_vector_type(16))) __bf16 v16bf;
typedef __attribute__((ext_vector_type(8)))  float  v8f;

union Frag {
    v16bf    v;
    uint32_t u[8];
    uint4    u4[2];
};

// round-to-nearest-even f32 -> bf16, packed pair
__device__ __forceinline__ uint32_t pk_bf16(float a, float b) {
    uint32_t ua = __float_as_uint(a), ub = __float_as_uint(b);
    ua = (ua + 0x7FFFu + ((ua >> 16) & 1u)) >> 16;
    ub = (ub + 0x7FFFu + ((ub >> 16) & 1u)) >> 16;
    return (ua & 0xFFFFu) | (ub << 16);
}

__device__ __forceinline__ float fast_tanh(float x) {
    float y;
    // CDNA5 transcendental; v_nop covers the TRANS-op result hazard
    asm volatile("v_tanh_f32 %0, %1\n\tv_nop" : "=v"(y) : "v"(x));
    return y;
}

__device__ __forceinline__ float fast_sigmoid(float x) {
    return 1.0f / (1.0f + __expf(-x));
}

// ---------------------------------------------------------------------------
// Prep 1: h_attn[b,d] = hidden[b,:] @ attn_W[0:512, d] + attn_b[d]   [128,512]
// ---------------------------------------------------------------------------
__global__ __launch_bounds__(256) void prep_hattn(
    const float* __restrict__ hidden, const float* __restrict__ attn_W,
    const float* __restrict__ attn_b, float* __restrict__ h_attn) {
    int t = blockIdx.x * 256 + threadIdx.x;      // 65536
    int d = t & 511, b = t >> 9;
    float acc = attn_b[d];
    const float* hp = hidden + b * 512;
    #pragma unroll 4
    for (int k = 0; k < 512; ++k) acc += hp[k] * attn_W[k * 512 + d];
    h_attn[b * 512 + d] = acc;
}

// ---------------------------------------------------------------------------
// Prep 2: gh[b,j] = hidden[b,:] @ W_hh[j,:]^T + b_hh[j]              [128,1536]
// ---------------------------------------------------------------------------
__global__ __launch_bounds__(256) void prep_gh(
    const float* __restrict__ hidden, const float* __restrict__ W_hh,
    const float* __restrict__ b_hh, float* __restrict__ gh) {
    int t = blockIdx.x * 256 + threadIdx.x;      // 196608
    int j = t % 1536, b = t / 1536;
    float acc = b_hh[j];
    const float* hp = hidden + b * 512;
    const float* wp = W_hh + (size_t)j * 512;
    #pragma unroll 4
    for (int k = 0; k < 512; ++k) acc += hp[k] * wp[k];
    gh[b * 1536 + j] = acc;
}

// ---------------------------------------------------------------------------
// Prep 3: gi0[b,j] = input[b,:256] @ W_ih[j,0:256]^T + b_ih[j]       [128,1536]
// ---------------------------------------------------------------------------
__global__ __launch_bounds__(256) void prep_gi0(
    const float* __restrict__ input, const float* __restrict__ W_ih,
    const float* __restrict__ b_ih, float* __restrict__ gi0) {
    int t = blockIdx.x * 256 + threadIdx.x;      // 196608
    int j = t % 1536, b = t / 1536;
    float acc = b_ih[j];
    const float* ip = input + b * 256;
    const float* wp = W_ih + (size_t)j * 768;    // cols [0,256)
    #pragma unroll 4
    for (int k = 0; k < 256; ++k) acc += ip[k] * wp[k];
    gi0[b * 1536 + j] = acc;
}

// ---------------------------------------------------------------------------
// Prep 4: out0[b,o] = input[b,:] @ out_W[o,1024:1280]^T + out_b[o]   [128,256]
// ---------------------------------------------------------------------------
__global__ __launch_bounds__(256) void prep_out0(
    const float* __restrict__ input, const float* __restrict__ out_W,
    const float* __restrict__ out_b, float* __restrict__ out0) {
    int t = blockIdx.x * 256 + threadIdx.x;      // 32768
    int o = t & 255, b = t >> 8;
    float acc = out_b[o];
    const float* ip = input + b * 256;
    const float* wp = out_W + (size_t)o * 1280 + 1024;
    #pragma unroll 4
    for (int k = 0; k < 256; ++k) acc += ip[k] * wp[k];
    out0[b * 256 + o] = acc;
}

// ---------------------------------------------------------------------------
// Prep 5: pack W_enc = attn_W[512:1024, :] into bf16 WMMA B-fragments.
// Layout: frag f = nt*16 + ks (nt in [0,32), ks in [0,16)); per frag 32 lanes
// x 16 bf16 (8 dwords) in the ISA 16-bit 32x16 register layout:
//   lane = half*16 + n,  dword j<4 : K = ks*32 + 2j   + 8*half (pair)
//                        dword j>=4: K = ks*32 + 16+2(j-4) + 8*half
// ---------------------------------------------------------------------------
__global__ __launch_bounds__(256) void prep_pack_wenc(
    const float* __restrict__ attn_W, uint32_t* __restrict__ wenc) {
    int t = blockIdx.x * 256 + threadIdx.x;      // 16384 = 512 frags * 32 lanes
    int lane = t & 31, frag = t >> 5;
    int nt = frag >> 4, ks = frag & 15;
    int n = nt * 16 + (lane & 15);
    int half = lane >> 4;
    uint32_t u[8];
    #pragma unroll
    for (int j = 0; j < 8; ++j) {
        int kcol = (j < 4 ? 2 * j : 16 + 2 * (j - 4)) + 8 * half;
        int k = ks * 32 + kcol;
        float f0 = attn_W[(size_t)(512 + k) * 512 + n];
        float f1 = attn_W[(size_t)(513 + k) * 512 + n];
        u[j] = pk_bf16(f0, f1);
    }
    uint4* dst = (uint4*)(wenc + ((size_t)frag * 32 + lane) * 8);
    dst[0] = make_uint4(u[0], u[1], u[2], u[3]);
    dst[1] = make_uint4(u[4], u[5], u[6], u[7]);
}

// ---------------------------------------------------------------------------
// Main kernel: fused  scores[b,s] = v . tanh(h_attn[b,:] + enc[s,b,:]@W_enc)
//
// One WG per PAIR of s values (grid 512). Both [128,512] enc slices are
// staged as bf16 in 256 KB of LDS (within the 320 KB/WG limit). Each of the
// 8 waves owns one 16-row b-tile in BOTH slices, so every B fragment fetched
// from L2 feeds two v_wmma_f32_16x16x32_bf16 ops (halves L2 B-traffic), and
// the h_attn bias + v loads are shared between the two slices. The bias tile
// is preloaded and used as the WMMA C-input, so the epilogue is just
// tanh + fma, with no dependent scalar-load chain.
// ---------------------------------------------------------------------------
__global__ __launch_bounds__(256) void attn_scores_kernel(
    const float* __restrict__ enc,        // [1024,128,512] f32
    const uint32_t* __restrict__ wenc,    // packed bf16 B-fragments
    const float* __restrict__ h_attn,     // [128,512]
    const float* __restrict__ v,          // [512]
    float* __restrict__ scores_T)         // [128,1024] (b-major)
{
    extern __shared__ uint32_t smem[];    // 256 x 512 bf16 = 65536 dwords
    const int s0  = blockIdx.x * 2;
    const int tid = threadIdx.x;

    // ---- phase 1: enc[s0], enc[s0+1] -> bf16 LDS (coalesced b128 loads) ----
    const float4* src = (const float4*)(enc + (size_t)s0 * 128 * 512);
    uint2* dst = (uint2*)smem;
    for (int i = tid; i < (256 * 512) / 4; i += 256) {
        float4 f = src[i];
        uint2 p;
        p.x = pk_bf16(f.x, f.y);
        p.y = pk_bf16(f.z, f.w);
        dst[i] = p;
    }
    __syncthreads();

    // ---- phase 2: WMMA sweep ----
    const int wave = tid >> 5;            // 0..7 -> b-tile base
    const int lane = tid & 31;
    const int half = lane >> 4;
    const int ln   = lane & 15;
    const int m0   = wave * 16;           // b base

    float sacc0[8], sacc1[8];
    #pragma unroll
    for (int i = 0; i < 8; ++i) { sacc0[i] = 0.0f; sacc1[i] = 0.0f; }

    const uint4* a4     = (const uint4*)smem;
    const int    abase0 = (m0 + ln) * 64 + half;          // slice s0
    const int    abase1 = (128 + m0 + ln) * 64 + half;    // slice s0+1

    for (int nt = 0; nt < 32; ++nt) {
        // batched bias tile load (C layout: lane -> N=ln, M = i + 8*half);
        // h_attn depends only on (b, n) so both s-slices share it
        float bias[8];
        const float* hb = h_attn + (size_t)(m0 + half * 8) * 512 + nt * 16 + ln;
        #pragma unroll
        for (int i = 0; i < 8; ++i) bias[i] = hb[(size_t)i * 512];

        v8f acc0, acc1;
        #pragma unroll
        for (int i = 0; i < 8; ++i) { acc0[i] = bias[i]; acc1[i] = bias[i]; }

        const uint4* b4 = ((const uint4*)wenc) + (size_t)nt * 1024 + lane * 2;
        #pragma unroll 4
        for (int ks = 0; ks < 16; ++ks) {
            Frag a0, a1, b;
            const int ai = ks * 4;
            a0.u4[0] = a4[abase0 + ai];       // ds_load_b128
            a0.u4[1] = a4[abase0 + ai + 2];   // ds_load_b128
            a1.u4[0] = a4[abase1 + ai];
            a1.u4[1] = a4[abase1 + ai + 2];
            b.u4[0]  = b4[0];                 // global_load_b128
            b.u4[1]  = b4[1];                 // global_load_b128
            __builtin_prefetch((const void*)(b4 + 64), 0, 1);
            b4 += 64;                         // next K-step fragment
            acc0 = __builtin_amdgcn_wmma_f32_16x16x32_bf16(
                false, a0.v, false, b.v, (short)0, acc0, false, false);
            acc1 = __builtin_amdgcn_wmma_f32_16x16x32_bf16(
                false, a1.v, false, b.v, (short)0, acc1, false, false);
        }
        // fused tanh + v-dot epilogue (bias already inside acc via C input)
        const float vv = v[nt * 16 + ln];
        #pragma unroll
        for (int i = 0; i < 8; ++i) {
            sacc0[i] += vv * fast_tanh(acc0[i]);
            sacc1[i] += vv * fast_tanh(acc1[i]);
        }
    }

    // ---- phase 3: reduce across the 16 lanes sharing each M ----
    #pragma unroll
    for (int i = 0; i < 8; ++i) {
        float x0 = sacc0[i], x1 = sacc1[i];
        x0 += __shfl_xor(x0, 1, 16);
        x0 += __shfl_xor(x0, 2, 16);
        x0 += __shfl_xor(x0, 4, 16);
        x0 += __shfl_xor(x0, 8, 16);
        x1 += __shfl_xor(x1, 1, 16);
        x1 += __shfl_xor(x1, 2, 16);
        x1 += __shfl_xor(x1, 4, 16);
        x1 += __shfl_xor(x1, 8, 16);
        sacc0[i] = x0;
        sacc1[i] = x1;
    }
    if (ln == 0) {
        #pragma unroll
        for (int i = 0; i < 8; ++i) {
            int b = m0 + half * 8 + i;
            scores_T[(size_t)b * 1024 + s0]     = sacc0[i];
            scores_T[(size_t)b * 1024 + s0 + 1] = sacc1[i];
        }
    }
}

// ---------------------------------------------------------------------------
// Softmax over s (per b) + context vector  weighted[b,:] = att . enc[:,b,:]
// ---------------------------------------------------------------------------
__global__ __launch_bounds__(256) void softmax_ctx_kernel(
    const float* __restrict__ scores_T,   // [128,1024]
    const float* __restrict__ enc,        // [1024,128,512]
    float* __restrict__ weighted)         // [128,512]
{
    __shared__ float att[1024];
    __shared__ float red[256];
    const int b = blockIdx.x, tid = threadIdx.x;

    float sc[4];
    float mx = -INFINITY;
    #pragma unroll
    for (int r = 0; r < 4; ++r) {
        sc[r] = scores_T[(size_t)b * 1024 + tid + r * 256];
        mx = fmaxf(mx, sc[r]);
    }
    red[tid] = mx; __syncthreads();
    for (int o = 128; o > 0; o >>= 1) {
        if (tid < o) red[tid] = fmaxf(red[tid], red[tid + o]);
        __syncthreads();
    }
    mx = red[0]; __syncthreads();

    float sum = 0.0f;
    #pragma unroll
    for (int r = 0; r < 4; ++r) { sc[r] = __expf(sc[r] - mx); sum += sc[r]; }
    red[tid] = sum; __syncthreads();
    for (int o = 128; o > 0; o >>= 1) {
        if (tid < o) red[tid] += red[tid + o];
        __syncthreads();
    }
    float inv = 1.0f / red[0];
    #pragma unroll
    for (int r = 0; r < 4; ++r) att[tid + r * 256] = sc[r] * inv;
    __syncthreads();

    // context: 2 floats per thread, coalesced b64 streams over enc
    const int e2 = tid;                   // float2 index, e = 2*tid
    float2 acc = make_float2(0.0f, 0.0f);
    const float2* ep = (const float2*)enc;
    for (int s = 0; s < 1024; ++s) {
        float w = att[s];
        float2 x = ep[(size_t)s * 32768 + b * 256 + e2];
        acc.x += w * x.x;
        acc.y += w * x.y;
    }
    ((float2*)weighted)[b * 256 + e2] = acc;
}

// ---------------------------------------------------------------------------
// GRU gates (torch order r,z,n) -> h_new  (written to d_out[32768:])
// ---------------------------------------------------------------------------
__global__ __launch_bounds__(256) void gru_kernel(
    const float* __restrict__ gi0, const float* __restrict__ gh,
    const float* __restrict__ weighted, const float* __restrict__ W_ih,
    const float* __restrict__ hidden, float* __restrict__ dout) {
    int t = blockIdx.x * 256 + threadIdx.x;       // 65536
    int h = t & 511, b = t >> 9;
    const float* wv = weighted + b * 512;
    const float* wr = W_ih + (size_t)h * 768 + 256;
    const float* wz = W_ih + (size_t)(512 + h) * 768 + 256;
    const float* wn = W_ih + (size_t)(1024 + h) * 768 + 256;
    float ar = 0.0f, az = 0.0f, an = 0.0f;
    #pragma unroll 4
    for (int k = 0; k < 512; ++k) {
        float w = wv[k];
        ar += w * wr[k];
        az += w * wz[k];
        an += w * wn[k];
    }
    const float* g0 = gi0 + b * 1536;
    const float* g1 = gh + b * 1536;
    float r = fast_sigmoid(g0[h] + ar + g1[h]);
    float z = fast_sigmoid(g0[512 + h] + az + g1[512 + h]);
    float n = fast_tanh(g0[1024 + h] + an + r * g1[1024 + h]);
    float hn = (1.0f - z) * n + z * hidden[b * 512 + h];
    dout[32768 + b * 512 + h] = hn;
}

// ---------------------------------------------------------------------------
// Output projection: out = out0 + h_new@out_W[:, :512]^T + weighted@[:,512:1024]^T
// ---------------------------------------------------------------------------
__global__ __launch_bounds__(256) void outproj_kernel(
    const float* __restrict__ out0, const float* __restrict__ weighted,
    const float* __restrict__ out_W, float* __restrict__ dout) {
    int t = blockIdx.x * 256 + threadIdx.x;       // 32768
    int o = t & 255, b = t >> 8;
    const float* wh = out_W + (size_t)o * 1280;
    const float* ww = wh + 512;
    const float* hn = dout + 32768 + b * 512;     // h_new written by gru_kernel
    const float* wv = weighted + b * 512;
    float acc = out0[b * 256 + o];
    #pragma unroll 4
    for (int k = 0; k < 512; ++k) acc += hn[k] * wh[k] + wv[k] * ww[k];
    dout[b * 256 + o] = acc;
}

// ---------------------------------------------------------------------------
extern "C" void kernel_launch(void* const* d_in, const int* in_sizes, int n_in,
                              void* d_out, int out_size, void* d_ws, size_t ws_size,
                              hipStream_t stream) {
    const float* input   = (const float*)d_in[0];
    const float* hidden  = (const float*)d_in[1];
    const float* enc     = (const float*)d_in[2];
    const float* attn_W  = (const float*)d_in[3];
    const float* attn_b  = (const float*)d_in[4];
    const float* v       = (const float*)d_in[5];
    const float* W_ih    = (const float*)d_in[6];
    const float* W_hh    = (const float*)d_in[7];
    const float* b_ih    = (const float*)d_in[8];
    const float* b_hh    = (const float*)d_in[9];
    const float* out_W   = (const float*)d_in[10];
    const float* out_b   = (const float*)d_in[11];
    float* dout = (float*)d_out;

    char* ws = (char*)d_ws;
    float*    h_attn   = (float*)(ws + 0);          // 128*512 f32   = 256 KB
    uint32_t* wenc     = (uint32_t*)(ws + 262144);  // 512 KB packed bf16
    float*    scores_T = (float*)(ws + 786432);     // 128*1024 f32  = 512 KB
    float*    weighted = (float*)(ws + 1310720);    // 128*512 f32   = 256 KB
    float*    gh       = (float*)(ws + 1572864);    // 128*1536 f32  = 768 KB
    float*    gi0      = (float*)(ws + 2359296);    // 128*1536 f32  = 768 KB
    float*    out0     = (float*)(ws + 3145728);    // 128*256 f32   = 128 KB

    // prep (independent of the big pass)
    prep_hattn<<<256, 256, 0, stream>>>(hidden, attn_W, attn_b, h_attn);
    prep_gh<<<768, 256, 0, stream>>>(hidden, W_hh, b_hh, gh);
    prep_gi0<<<768, 256, 0, stream>>>(input, W_ih, b_ih, gi0);
    prep_out0<<<128, 256, 0, stream>>>(input, out_W, out_b, out0);
    prep_pack_wenc<<<64, 256, 0, stream>>>(attn_W, wenc);

    // fused attention-energy GEMM (bf16 WMMA) + tanh + v-dot, 2 s-slices/WG
    attn_scores_kernel<<<512, 256, 262144, stream>>>(enc, wenc, h_attn, v, scores_T);

    // softmax over s + context vector
    softmax_ctx_kernel<<<128, 256, 0, stream>>>(scores_T, enc, weighted);

    // GRU step and output projection
    gru_kernel<<<256, 256, 0, stream>>>(gi0, gh, weighted, W_ih, hidden, dout);
    outproj_kernel<<<128, 256, 0, stream>>>(out0, weighted, out_W, dout);
}
